// Model_39676907887682
// MI455X (gfx1250) — matrix operand order repaired and checked by reading, not compiled, not run on hardware
//
#include <hip/hip_runtime.h>
#include <stdint.h>

// ---- dims from the reference ----
#define NB   4
#define NH   16
#define SQL  1024
#define SKL  1024
#define DD   64
#define DVV  64

typedef __attribute__((ext_vector_type(16))) __bf16 v16bf;
typedef __attribute__((ext_vector_type(4)))  __bf16 bf16x4;
typedef __attribute__((ext_vector_type(8)))  float  v8f;
typedef __attribute__((ext_vector_type(4)))  unsigned int u32x4;

union FragBF16 {            // 16 bf16 = 32B = two 16B LDS loads
    v16bf v;
    u32x4 q[2];
};

__device__ __forceinline__ uint32_t mix32(uint32_t x) {
    x ^= x >> 16; x *= 0x7feb352dU;
    x ^= x >> 15; x *= 0x846ca68bU;
    x ^= x >> 16;
    return x;
}

#define BM 64                 // Q rows per block (4 waves x 16)
#define BN 64                 // K rows per tile
#define LDW (DD + 8)          // 72 halfwords = 144B row stride (16B aligned)
#define KEEP8 51u             // ~0.2 * 256 (per-byte dropout threshold)
#define NKB (SKL / BN)

__launch_bounds__(128, 2)
__global__ void fa_fwd_kernel(const float* __restrict__ Q,
                              const float* __restrict__ K,
                              const float* __restrict__ V,
                              const float* __restrict__ ISF,
                              float* __restrict__ O)
{
    // raw f32 staging filled by async HBM->LDS DMA (one contiguous 16KB tile each)
    __shared__ __align__(16) float  Kraw[BN * DD];
    __shared__ __align__(16) float  Vraw[BN * DVV];
    // bf16 operand tiles
    __shared__ __align__(16) __bf16 Ks [BN * LDW];        // K tile, row-major [k][d]
    __shared__ __align__(16) __bf16 VTs[DVV * LDW];       // V tile transposed [dv][k]
    __shared__ __align__(16) __bf16 Ps [4 * 16 * LDW];    // per-wave P strips [m][k]

    const int tid   = threadIdx.x;
    const int wave  = tid >> 5;
    const int lane  = tid & 31;
    const int lhalf = lane >> 4;     // 0: lanes 0-15, 1: lanes 16-31
    const int l16   = lane & 15;

    const int bh    = blockIdx.y;                    // fused batch*head
    const int qbase = blockIdx.x * BM;
    const int qrowA = qbase + wave * 16 + l16;       // row for A-fragments (M = lane%16)
    const int qrowC = qbase + wave * 16 + lhalf * 8; // +v for C-layout rows

    const float* Qp = Q + (size_t)bh * SQL * DD;
    const float* Kp = K + (size_t)bh * SKL * DD;
    const float* Vp = V + (size_t)bh * SKL * DVV;
    float*       Op = O + (size_t)bh * SQL * DVV;

    const uint32_t kraw_lds = (uint32_t)(uintptr_t)&Kraw[0];
    const uint32_t vraw_lds = (uint32_t)(uintptr_t)&Vraw[0];

    // ---- load this wave's Q fragment once (16x64 strip -> 2 chunks of K=32) ----
    FragBF16 aq[2];
#pragma unroll
    for (int c = 0; c < 2; ++c) {
#pragma unroll
        for (int g = 0; g < 2; ++g) {
            // halfwords g*8 .. g*8+7 hold K = c*32 + lhalf*16 + g*8 + (0..7)
            const float* src = Qp + (size_t)qrowA * DD + c * 32 + lhalf * 16 + g * 8;
            float4 f0 = ((const float4*)src)[0];
            float4 f1 = ((const float4*)src)[1];
            aq[c].v[g * 8 + 0] = (__bf16)f0.x;  aq[c].v[g * 8 + 1] = (__bf16)f0.y;
            aq[c].v[g * 8 + 2] = (__bf16)f0.z;  aq[c].v[g * 8 + 3] = (__bf16)f0.w;
            aq[c].v[g * 8 + 4] = (__bf16)f1.x;  aq[c].v[g * 8 + 5] = (__bf16)f1.y;
            aq[c].v[g * 8 + 6] = (__bf16)f1.z;  aq[c].v[g * 8 + 7] = (__bf16)f1.w;
        }
    }

    // ---- async DMA of tile 0 (contiguous 16KB per tile, 16B per lane per issue) ----
#pragma unroll
    for (int e = 0; e < 8; ++e) {
        const int f = e * 128 + tid;                       // float4 index 0..1023
        const uint32_t lk = kraw_lds + (uint32_t)f * 16u;
        const uint32_t lv = vraw_lds + (uint32_t)f * 16u;
        const uint64_t ak = (uint64_t)(uintptr_t)(Kp + (size_t)f * 4);
        const uint64_t av = (uint64_t)(uintptr_t)(Vp + (size_t)f * 4);
        asm volatile("global_load_async_to_lds_b128 %0, %1, off"
                     :: "v"(lk), "v"(ak) : "memory");
        asm volatile("global_load_async_to_lds_b128 %0, %1, off"
                     :: "v"(lv), "v"(av) : "memory");
    }
    asm volatile("s_wait_asynccnt 0x0" ::: "memory");
    __syncthreads();

    // ---- online-softmax state (C layout: row = v + 8*lhalf, col = l16) ----
    float m_i[8], l_i[8];
    v8f   o_acc[4];
#pragma unroll
    for (int v = 0; v < 8; ++v) { m_i[v] = -1e30f; l_i[v] = 0.f; }
#pragma unroll
    for (int j = 0; j < 4; ++j)
#pragma unroll
        for (int e = 0; e < 8; ++e) o_acc[j][e] = 0.f;

    for (int kb = 0; kb < NKB; ++kb) {
        const int kbase = kb * BN;

        // ---- convert raw f32 tile (kb) -> bf16 Ks / transposed VTs ----
#pragma unroll
        for (int e = 0; e < 8; ++e) {
            const int g  = e * 128 + tid;      // 0..1023
            const int r  = g >> 4;
            const int c4 = (g & 15) * 4;
            const float4 f = *(const float4*)&Kraw[g * 4];
            bf16x4 b;
            b[0] = (__bf16)f.x; b[1] = (__bf16)f.y;
            b[2] = (__bf16)f.z; b[3] = (__bf16)f.w;
            *(bf16x4*)&Ks[r * LDW + c4] = b;
        }
#pragma unroll
        for (int e = 0; e < 8; ++e) {
            const int g  = e * 128 + tid;      // 0..1023
            const int c  = g & 63;             // dv column
            const int r4 = (g >> 6) * 4;       // k row group
            bf16x4 b;
            b[0] = (__bf16)Vraw[(r4 + 0) * DVV + c];
            b[1] = (__bf16)Vraw[(r4 + 1) * DVV + c];
            b[2] = (__bf16)Vraw[(r4 + 2) * DVV + c];
            b[3] = (__bf16)Vraw[(r4 + 3) * DVV + c];
            *(bf16x4*)&VTs[c * LDW + r4] = b;
        }
        __syncthreads();   // Ks/VTs ready; raw buffers free for the next async fill

        // ---- kick off async DMA for tile kb+1; overlaps with compute below ----
        if (kb + 1 < NKB) {
            const float* gK = Kp + (size_t)(kbase + BN) * DD;
            const float* gV = Vp + (size_t)(kbase + BN) * DVV;
#pragma unroll
            for (int e = 0; e < 8; ++e) {
                const int f = e * 128 + tid;
                const uint32_t lk = kraw_lds + (uint32_t)f * 16u;
                const uint32_t lv = vraw_lds + (uint32_t)f * 16u;
                const uint64_t ak = (uint64_t)(uintptr_t)(gK + (size_t)f * 4);
                const uint64_t av = (uint64_t)(uintptr_t)(gV + (size_t)f * 4);
                asm volatile("global_load_async_to_lds_b128 %0, %1, off"
                             :: "v"(lk), "v"(ak) : "memory");
                asm volatile("global_load_async_to_lds_b128 %0, %1, off"
                             :: "v"(lv), "v"(av) : "memory");
            }
        }

        // ---- S = Q K^T for this wave's 16x64 strip ----
        v8f s[4];
#pragma unroll
        for (int j = 0; j < 4; ++j) {
            FragBF16 bk[2];
#pragma unroll
            for (int c = 0; c < 2; ++c) {
                // B layout: N = l16 (key row j*16+l16), K = c*32 + lhalf*16 + (0..15)
                const __bf16* p = &Ks[(j * 16 + l16) * LDW + c * 32 + lhalf * 16];
                bk[c].q[0] = *(const u32x4*)p;
                bk[c].q[1] = *(const u32x4*)(p + 8);
            }
            v8f acc = {0.f,0.f,0.f,0.f,0.f,0.f,0.f,0.f};
            acc = __builtin_amdgcn_wmma_f32_16x16x32_bf16(false, aq[0].v, false, bk[0].v,
                                                          (short)0, acc, false, false);
            acc = __builtin_amdgcn_wmma_f32_16x16x32_bf16(false, aq[1].v, false, bk[1].v,
                                                          (short)0, acc, false, false);
            s[j] = acc;
        }

        // ---- scale by rcp(ISF) (L2-resident), track row max ----
        float cmax[8];
#pragma unroll
        for (int v = 0; v < 8; ++v) cmax[v] = -1e30f;
#pragma unroll
        for (int j = 0; j < 4; ++j) {
            const int kg = kbase + j * 16 + l16;
#pragma unroll
            for (int v = 0; v < 8; ++v) {
                const int qg = qrowC + v;
                const float sv = s[j][v] * __builtin_amdgcn_rcpf(ISF[(size_t)qg * SKL + kg]);
                s[j][v] = sv;
                cmax[v] = fmaxf(cmax[v], sv);
            }
        }
        // reduce max across the 16 lanes holding one row
#pragma unroll
        for (int v = 0; v < 8; ++v) {
            float x = cmax[v];
#pragma unroll
            for (int off = 1; off < 16; off <<= 1)
                x = fmaxf(x, __shfl_xor(x, off, 32));
            cmax[v] = x;
        }

        float corr[8];
#pragma unroll
        for (int v = 0; v < 8; ++v) {
            const float mn = fmaxf(m_i[v], cmax[v]);
            corr[v] = __expf(m_i[v] - mn);
            m_i[v]  = mn;
        }

        // one hash per row per k-block; byte j gates column tile j
        uint32_t hh[8];
#pragma unroll
        for (int v = 0; v < 8; ++v)
            hh[v] = mix32(((uint32_t)bh << 26) ^ ((uint32_t)(qrowC + v) << 10)
                          ^ (uint32_t)(kbase + l16));

        // ---- p = exp(s - m); l += rowsum(p) (pre-dropout); dropout -> Ps ----
        float rsum[8];
#pragma unroll
        for (int v = 0; v < 8; ++v) rsum[v] = 0.f;
#pragma unroll
        for (int j = 0; j < 4; ++j) {
#pragma unroll
            for (int v = 0; v < 8; ++v) {
                const float p = __expf(s[j][v] - m_i[v]);
                rsum[v] += p;
                const float pd = (((hh[v] >> (8 * j)) & 0xFFu) < KEEP8) ? p : 0.f;
                Ps[(wave * 16 + lhalf * 8 + v) * LDW + j * 16 + l16] = (__bf16)pd;
            }
        }
#pragma unroll
        for (int v = 0; v < 8; ++v) {
            float x = rsum[v];
#pragma unroll
            for (int off = 1; off < 16; off <<= 1)
                x += __shfl_xor(x, off, 32);
            l_i[v] = l_i[v] * corr[v] + x;
        }
#pragma unroll
        for (int j = 0; j < 4; ++j)
#pragma unroll
            for (int v = 0; v < 8; ++v) o_acc[j][v] *= corr[v];

        __syncthreads();   // Ps strips visible (C-layout -> A-layout transpose)

        // ---- O += P V : A = Ps strip (16xBN), B = VT tiles ----
#pragma unroll
        for (int c = 0; c < 2; ++c) {
            FragBF16 ap;
            // A layout: halfwords 0-15 at K = c*32+lhalf*16 + (0..15)
            const __bf16* pp = &Ps[(wave * 16 + l16) * LDW + c * 32 + lhalf * 16];
            ap.q[0] = *(const u32x4*)pp;
            ap.q[1] = *(const u32x4*)(pp + 8);
#pragma unroll
            for (int j = 0; j < 4; ++j) {
                FragBF16 bv;
                const __bf16* vp = &VTs[(j * 16 + l16) * LDW + c * 32 + lhalf * 16];
                bv.q[0] = *(const u32x4*)vp;
                bv.q[1] = *(const u32x4*)(vp + 8);
                o_acc[j] = __builtin_amdgcn_wmma_f32_16x16x32_bf16(
                    false, ap.v, false, bv.v, (short)0, o_acc[j], false, false);
            }
        }

        // ---- tile kb+1 has been streaming in during compute; close the iteration ----
        asm volatile("s_wait_asynccnt 0x0" ::: "memory");
        __syncthreads();
    }

    // ---- epilogue: out = o * (1 / (l * keep)) = o * 5 * rcp(l) ----
#pragma unroll
    for (int v = 0; v < 8; ++v) {
        const int qg = qrowC + v;
        const float invl = 5.0f * __builtin_amdgcn_rcpf(l_i[v]);
#pragma unroll
        for (int j = 0; j < 4; ++j) {
            Op[(size_t)qg * DVV + j * 16 + l16] = o_acc[j][v] * invl;
        }
    }
}

extern "C" void kernel_launch(void* const* d_in, const int* in_sizes, int n_in,
                              void* d_out, int out_size, void* d_ws, size_t ws_size,
                              hipStream_t stream) {
    // setup_inputs order: x2 (unused), query, key_w, value, inv_scale_factor
    const float* Q   = (const float*)d_in[1];
    const float* K   = (const float*)d_in[2];
    const float* V   = (const float*)d_in[3];
    const float* ISF = (const float*)d_in[4];
    float* O = (float*)d_out;

    dim3 grid(SQL / BM, NB * NH);
    fa_fwd_kernel<<<grid, 128, 0, stream>>>(Q, K, V, ISF, O);
    (void)in_sizes; (void)n_in; (void)out_size; (void)d_ws; (void)ws_size;
}
